// Qwen3SeerAttention_64604898066601
// MI455X (gfx1250) — matrix operand order repaired and checked by reading, not compiled, not run on hardware
//
#include <hip/hip_runtime.h>
#include <hip/hip_bf16.h>
#include <cstdint>
#include <cstddef>

// ---------------- problem constants (match reference) ----------------
constexpr int B_   = 2;
constexpr int S_   = 2048;
constexpr int HID_ = 2048;
constexpr int H_   = 16;
constexpr int HKV_ = 8;
constexpr int D_   = 128;
constexpr int GB_  = 64;
constexpr int NB_  = S_ / GB_;   // 32
constexpr int G_   = H_ / HKV_;  // 2
constexpr int BUDGET_ = 8;       // 512/64
constexpr int WINDOW_ = 4;       // 256/64
#define SCALE_F 0.08838834764831845f
#define NEG_F   (-1.0e30f)
#define EPS_F   1e-6f

typedef __attribute__((ext_vector_type(16))) __bf16 v16bf;
typedef __attribute__((ext_vector_type(8)))  float  v8f;

__device__ __forceinline__ __bf16 f2bf(float f) { return (__bf16)f; }
__device__ __forceinline__ float  bf2f(__bf16 b) { return (float)b; }

// ---------------- CDNA5 async global->LDS staging (guarded) ----------------
#if defined(__gfx1250__) && __has_builtin(__builtin_amdgcn_global_load_async_to_lds_b128)
#define ASYNC_LDS 1
#else
#define ASYNC_LDS 0
#endif

// Param type per hipcc diagnostic: pointer to GCC-style 16-byte int vector,
// global (AS1 / __device__) source first, LDS (AS3) destination second.
typedef int v4i_vs __attribute__((vector_size(16)));

__device__ __forceinline__ void copy16_g2l(const __bf16* g, __bf16* l) {
#if ASYNC_LDS
  __builtin_amdgcn_global_load_async_to_lds_b128(
      (__attribute__((address_space(1))) v4i_vs*)(uintptr_t)g,
      (__attribute__((address_space(3))) v4i_vs*)(uint32_t)(uintptr_t)l,
      0, 0);
#else
  *reinterpret_cast<uint4*>(l) = *reinterpret_cast<const uint4*>(g);
#endif
}
__device__ __forceinline__ void g2l_wait() {
#if ASYNC_LDS
  asm volatile("s_wait_asynccnt 0x0" ::: "memory");
#endif
}

// ---------------- fp32 -> bf16 cast ----------------
__global__ void cast_bf16_kernel(const float* __restrict__ x, __bf16* __restrict__ y, size_t n) {
  size_t i = (size_t)blockIdx.x * blockDim.x + threadIdx.x;
  size_t stride = (size_t)gridDim.x * blockDim.x;
  for (; i < n; i += stride) y[i] = f2bf(x[i]);
}

// ---------------- WMMA GEMM: C[M,N] = A[M,K] (bf16) @ W[K,N] (bf16), fp32 out ----
// 256 threads = 8 waves; 128x64 tile; each wave owns a 32x32 block (4 wmma tiles).
__global__ __launch_bounds__(256) void gemm_bf16_kernel(
    const __bf16* __restrict__ A, const __bf16* __restrict__ W,
    float* __restrict__ C, int M, int N, int K) {
  __shared__ __align__(16) __bf16 lA[128 * 32];   // [row][k]   8 KB
  __shared__ __align__(16) __bf16 lB[64 * 32];    // [n][k]     4 KB (transposed)
  const int tid  = threadIdx.x;
  const int lane = tid & 31;
  const int wid  = tid >> 5;
  const int hl   = lane >> 4;
  const int r    = lane & 15;
  const int wm   = wid >> 1;       // 0..3 : 32-row group
  const int wn   = wid & 1;        // 0..1 : 32-col group
  const int m0   = blockIdx.y * 128;
  const int n0   = blockIdx.x * 64;

  const v8f vzero = {0.f,0.f,0.f,0.f,0.f,0.f,0.f,0.f};
  v8f acc[2][2];
#pragma unroll
  for (int i = 0; i < 2; ++i)
#pragma unroll
    for (int j = 0; j < 2; ++j) acc[i][j] = vzero;

  for (int k0 = 0; k0 < K; k0 += 32) {
    // stage A 128x32 (two b128 per thread, async if available)
    {
      const int row = tid >> 1, cb = (tid & 1) * 16;
      const __bf16* ga = &A[(size_t)(m0 + row) * K + k0 + cb];
      __bf16* la = &lA[row * 32 + cb];
      copy16_g2l(ga, la);
      copy16_g2l(ga + 8, la + 8);
    }
    // stage W 32x64 transposed into lB[n][k]
    {
      const int kk = tid >> 3, nn = (tid & 7) * 8;
      uint4 v = *reinterpret_cast<const uint4*>(&W[(size_t)(k0 + kk) * N + n0 + nn]);
      const __bf16* e = reinterpret_cast<const __bf16*>(&v);
#pragma unroll
      for (int j = 0; j < 8; ++j) lB[(nn + j) * 32 + kk] = e[j];
      if (k0 + 32 < K)
        __builtin_prefetch(&W[(size_t)(k0 + 32 + kk) * N + n0 + nn], 0, 3);
    }
    g2l_wait();
    __syncthreads();

    // A fragments (contiguous k = 16*hl + s), B fragments, 4 WMMAs
    v16bf af[2], bfr[2];
    const int kb0 = hl * 16, kb1 = hl * 16 + 8;
#pragma unroll
    for (int i = 0; i < 2; ++i) {
      const __bf16* ap = &lA[(wm * 32 + i * 16 + r) * 32];
#pragma unroll
      for (int j = 0; j < 8; ++j) { af[i][j] = ap[kb0 + j]; af[i][8 + j] = ap[kb1 + j]; }
    }
#pragma unroll
    for (int j = 0; j < 2; ++j) {
      const __bf16* bp = &lB[(wn * 32 + j * 16 + r) * 32 + hl * 16];
#pragma unroll
      for (int q = 0; q < 16; ++q) bfr[j][q] = bp[q];
    }
#pragma unroll
    for (int i = 0; i < 2; ++i)
#pragma unroll
      for (int j = 0; j < 2; ++j)
        acc[i][j] = __builtin_amdgcn_wmma_f32_16x16x32_bf16(false, af[i], false, bfr[j], (short)0, acc[i][j], false, false);
    __syncthreads();
  }

  // store (C/D layout: elem e -> row hl*8+e, col = lane&15 in tile)
#pragma unroll
  for (int i = 0; i < 2; ++i) {
#pragma unroll
    for (int j = 0; j < 2; ++j) {
      const int col = n0 + wn * 32 + j * 16 + r;
#pragma unroll
      for (int e = 0; e < 8; ++e) {
        const int row = m0 + wm * 32 + i * 16 + hl * 8 + e;
        C[(size_t)row * N + col] = acc[i][j][e];
      }
    }
  }
}

// ---------------- fused RMSNorm + RoPE, fp32 in -> bf16 out ----------------
__global__ __launch_bounds__(128) void normrope_kernel(
    const float* __restrict__ X, const float* __restrict__ w,
    const float* __restrict__ cosb, const float* __restrict__ sinb,
    __bf16* __restrict__ Y, int nh) {
  const int th = blockIdx.x;
  const int bs = th / nh;          // b*S + s
  const int h  = th % nh;
  const int s  = bs % S_;
  const int d  = threadIdx.x;
  const size_t base = ((size_t)bs * nh + h) * D_;

  float x = X[base + d];
  float ss = x * x;
#pragma unroll
  for (int m = 16; m >= 1; m >>= 1) ss += __shfl_xor(ss, m, 32);
  __shared__ float red[4];
  const int lane = d & 31, wv = d >> 5;
  if (lane == 0) red[wv] = ss;
  __syncthreads();
  const float tot = red[0] + red[1] + red[2] + red[3];
  const float rms = rsqrtf(tot / (float)D_ + EPS_F);

  const float y  = w[d] * x * rms;
  const int   dp = (d < 64) ? d + 64 : d - 64;
  const float yp = w[dp] * X[base + dp] * rms;
  const float c  = cosb[s * D_ + d];
  const float sn = sinb[s * D_ + d];
  const float out = (d < 64) ? (y * c - yp * sn) : (y * c + yp * sn);
  Y[base + d] = f2bf(out);
}

// ---------------- block pooling of post-rope q/k ----------------
__global__ __launch_bounds__(128) void pool_kernel(
    const __bf16* __restrict__ q, const __bf16* __restrict__ k,
    float* __restrict__ qp, float* __restrict__ kp) {
  const int idx = blockIdx.x;            // ((b*NB + nb)*HKV + hk)
  const int hk = idx % HKV_;
  const int nb = (idx / HKV_) % NB_;
  const int b  = idx / (HKV_ * NB_);
  const int d  = threadIdx.x;
  float sq = 0.f, sk = 0.f;
  for (int t = 0; t < GB_; ++t) {
    const int s = nb * GB_ + t;
    const size_t qb = (((size_t)b * S_ + s) * H_ + hk * G_) * D_ + d;
    sq += bf2f(q[qb]) + bf2f(q[qb + D_]);
    const size_t kb = (((size_t)b * S_ + s) * HKV_ + hk) * D_ + d;
    sk += bf2f(k[kb]);
  }
  qp[(size_t)idx * D_ + d] = sq / (float)(GB_ * G_);
  kp[(size_t)idx * D_ + d] = sk / (float)GB_;
}

// ---------------- block score + top-k + window mask ----------------
__global__ __launch_bounds__(32) void blockmask_kernel(
    const float* __restrict__ qp, const float* __restrict__ kp, int* __restrict__ mask) {
  const int idx = blockIdx.x;            // (b*HKV + hk)*NB + qb
  const int qb = idx % NB_;
  const int hk = (idx / NB_) % HKV_;
  const int b  = idx / (NB_ * HKV_);
  const int kb = threadIdx.x;

  const float* qv = &qp[(((size_t)b * NB_ + qb) * HKV_ + hk) * D_];
  const float* kv = &kp[(((size_t)b * NB_ + kb) * HKV_ + hk) * D_];
  float sc = 0.f;
  for (int d = 0; d < D_; ++d) sc += qv[d] * kv[d];
  sc *= SCALE_F;
  const bool causal = (kb <= qb);
  const float s = causal ? sc : NEG_F;
  int cnt = 0;
  for (int j = 0; j < NB_; ++j) {
    float sj = __shfl(s, j, 32);
    if (sj > s) cnt++;
  }
  const bool sel    = cnt < BUDGET_;                 // s >= 8th-largest
  const bool window = causal && (qb - kb < WINDOW_);
  mask[(size_t)idx * NB_ + kb] = ((sel || window) && causal) ? 1 : 0;
}

// ---------------- flash attention over selected 64-wide k-blocks ----------------
// 4 waves per block; block = (b, h, 64-row q block); K/V staged in LDS.
__global__ __launch_bounds__(128) void attn_wmma_kernel(
    const __bf16* __restrict__ Q,   // [B,S,H,D]
    const __bf16* __restrict__ Km,  // [B,S,HKV,D]
    const __bf16* __restrict__ Vm,  // [B,S,HKV,D]
    const int* __restrict__ mask,   // [B,HKV,NB,NB]
    __bf16* __restrict__ O) {       // [B,S,H,D]
  __shared__ __align__(16) __bf16 lK[64 * 128];    // [kk][d]  16 KB
  __shared__ __align__(16) __bf16 lVt[128 * 64];   // [d][kk]  16 KB
  __shared__ __align__(16) __bf16 lP[4][16 * 64];  // per-wave P, 8 KB

  const int idx = blockIdx.x;           // (b*H + h)*NB + qb
  const int qb = idx % NB_;
  const int h  = (idx / NB_) % H_;
  const int b  = idx / (NB_ * H_);
  const int hk = h / G_;
  const int tid = threadIdx.x;
  const int wid = tid >> 5;
  const int lane = tid & 31;
  const int hl = lane >> 4, r = lane & 15;
  const int q0 = qb * 64 + wid * 16;    // 16 q rows per wave

  // Q fragments (4 over d, contiguous k = 16*hl + j)
  v16bf qf[4];
  {
    const size_t qbase = (((size_t)b * S_ + q0 + r) * H_ + h) * D_;
#pragma unroll
    for (int kd = 0; kd < 4; ++kd) {
      const int o0 = kd * 32 + hl * 16;
      const int o1 = kd * 32 + hl * 16 + 8;
#pragma unroll
      for (int j = 0; j < 8; ++j) { qf[kd][j] = Q[qbase + o0 + j]; qf[kd][8 + j] = Q[qbase + o1 + j]; }
    }
  }

  const v8f vzero = {0.f,0.f,0.f,0.f,0.f,0.f,0.f,0.f};
  float mrow[8], lrow[8];
  v8f o_acc[8];
#pragma unroll
  for (int e = 0; e < 8; ++e) { mrow[e] = -3.0e38f; lrow[e] = 0.f; }
#pragma unroll
  for (int dt = 0; dt < 8; ++dt) o_acc[dt] = vzero;

  const int* mrowp = &mask[(((size_t)b * HKV_ + hk) * NB_ + qb) * NB_];

  for (int kb = 0; kb <= qb; ++kb) {
    if (!mrowp[kb]) continue;           // block-uniform
    __syncthreads();                    // previous iteration's LDS reads done

    // ---- stage K (row-major, async) and V (transposed, sync) ----
    {
      const int row = tid >> 1;         // 0..63
      const int c0  = (tid & 1) * 64;   // 0 / 64
      const size_t gbase = (((size_t)b * S_ + kb * GB_ + row) * HKV_ + hk) * D_ + c0;
      const __bf16* gk = &Km[gbase];
      __bf16* lk = &lK[row * 128 + c0];
#pragma unroll
      for (int j = 0; j < 8; ++j) copy16_g2l(gk + j * 8, lk + j * 8);
      const __bf16* gv = &Vm[gbase];
#pragma unroll
      for (int j = 0; j < 8; ++j) {
        uint4 vv = *reinterpret_cast<const uint4*>(gv + j * 8);
        const __bf16* e = reinterpret_cast<const __bf16*>(&vv);
#pragma unroll
        for (int q = 0; q < 8; ++q) lVt[(c0 + j * 8 + q) * 64 + row] = e[q];
      }
      g2l_wait();
    }
    __syncthreads();

    // ---- S = Q K^T : 4 tiles of 16x16 per wave ----
    v8f st[4];
#pragma unroll
    for (int nt = 0; nt < 4; ++nt) {
      v8f stv = vzero;
#pragma unroll
      for (int kd = 0; kd < 4; ++kd) {
        v16bf bfr;
        const __bf16* bp = &lK[(nt * 16 + r) * 128 + kd * 32 + hl * 16];
#pragma unroll
        for (int j = 0; j < 16; ++j) bfr[j] = bp[j];
        stv = __builtin_amdgcn_wmma_f32_16x16x32_bf16(false, qf[kd], false, bfr, (short)0, stv, false, false);
      }
      st[nt] = stv;
    }

    // ---- scale + per-element causal, row max ----
    float mnew[8];
#pragma unroll
    for (int e = 0; e < 8; ++e) {
      const int row = q0 + hl * 8 + e;
      float mx = mrow[e];
#pragma unroll
      for (int nt = 0; nt < 4; ++nt) {
        float v = st[nt][e] * SCALE_F;
        const int col = kb * GB_ + nt * 16 + r;
        if (col > row) v = NEG_F;
        st[nt][e] = v;
        mx = fmaxf(mx, v);
      }
#pragma unroll
      for (int m = 1; m < 16; m <<= 1) mx = fmaxf(mx, __shfl_xor(mx, m, 32));
      mnew[e] = mx;
    }

    // ---- online softmax update ----
#pragma unroll
    for (int e = 0; e < 8; ++e) {
      const float alpha = __expf(mrow[e] - mnew[e]);
      mrow[e] = mnew[e];
      float psum = 0.f;
#pragma unroll
      for (int nt = 0; nt < 4; ++nt) {
        float p = __expf(st[nt][e] - mnew[e]);
        st[nt][e] = p;
        psum += p;
      }
#pragma unroll
      for (int m = 1; m < 16; m <<= 1) psum += __shfl_xor(psum, m, 32);
      lrow[e] = lrow[e] * alpha + psum;
#pragma unroll
      for (int dt = 0; dt < 8; ++dt) o_acc[dt][e] *= alpha;
    }

    // ---- repack P (C layout -> A layout) via per-wave LDS ----
#pragma unroll
    for (int nt = 0; nt < 4; ++nt)
#pragma unroll
      for (int e = 0; e < 8; ++e)
        lP[wid][(hl * 8 + e) * 64 + nt * 16 + r] = f2bf(st[nt][e]);
    __syncthreads();

    v16bf pf[2];
#pragma unroll
    for (int kt = 0; kt < 2; ++kt) {
      const __bf16* pp = &lP[wid][r * 64 + kt * 32];
      const int o0 = hl * 16, o1 = hl * 16 + 8;
#pragma unroll
      for (int j = 0; j < 8; ++j) { pf[kt][j] = pp[o0 + j]; pf[kt][8 + j] = pp[o1 + j]; }
    }

    // ---- O += P @ V (B fragments from transposed V in LDS) ----
#pragma unroll
    for (int dt = 0; dt < 8; ++dt) {
#pragma unroll
      for (int kt = 0; kt < 2; ++kt) {
        v16bf vfr;
        const __bf16* vp = &lVt[(dt * 16 + r) * 64 + kt * 32 + hl * 16];
#pragma unroll
        for (int j = 0; j < 16; ++j) vfr[j] = vp[j];
        o_acc[dt] = __builtin_amdgcn_wmma_f32_16x16x32_bf16(false, pf[kt], false, vfr, (short)0, o_acc[dt], false, false);
      }
    }
  }

  // ---- finalize: /= l, store bf16 [b,s,h,d] ----
#pragma unroll
  for (int dt = 0; dt < 8; ++dt) {
#pragma unroll
    for (int e = 0; e < 8; ++e) {
      const int row = q0 + hl * 8 + e;
      const float val = o_acc[dt][e] / lrow[e];
      O[(((size_t)b * S_ + row) * H_ + h) * D_ + dt * 16 + r] = f2bf(val);
    }
  }
}

// ---------------- host launch ----------------
extern "C" void kernel_launch(void* const* d_in, const int* in_sizes, int n_in,
                              void* d_out, int out_size, void* d_ws, size_t ws_size,
                              hipStream_t stream) {
  const float* hidden = (const float*)d_in[0];
  const float* cosb   = (const float*)d_in[1];
  const float* sinb   = (const float*)d_in[2];
  const float* Wq     = (const float*)d_in[3];
  const float* Wk     = (const float*)d_in[4];
  const float* Wv     = (const float*)d_in[5];
  const float* Wo     = (const float*)d_in[6];
  const float* qn     = (const float*)d_in[7];
  const float* kn     = (const float*)d_in[8];
  float* out = (float*)d_out;

  const size_t M = (size_t)B_ * S_;         // 4096
  size_t off = 0;
  auto alloc = [&](size_t bytes) -> void* {
    void* p = (char*)d_ws + off;
    off += (bytes + 255) & ~(size_t)255;
    return p;
  };
  __bf16* hbf = (__bf16*)alloc(M * HID_ * 2);
  __bf16* wqb = (__bf16*)alloc((size_t)HID_ * H_ * D_ * 2);
  __bf16* wkb = (__bf16*)alloc((size_t)HID_ * HKV_ * D_ * 2);
  __bf16* wvb = (__bf16*)alloc((size_t)HID_ * HKV_ * D_ * 2);
  __bf16* wob = (__bf16*)alloc((size_t)H_ * D_ * HID_ * 2);
  float*  Qf  = (float*)alloc(M * H_ * D_ * 4);
  float*  Kf  = (float*)alloc(M * HKV_ * D_ * 4);
  float*  Vf  = (float*)alloc(M * HKV_ * D_ * 4);
  __bf16* qbf = (__bf16*)alloc(M * H_ * D_ * 2);
  __bf16* kbf = (__bf16*)alloc(M * HKV_ * D_ * 2);
  __bf16* vbf = (__bf16*)alloc(M * HKV_ * D_ * 2);
  float*  qp  = (float*)alloc((size_t)B_ * NB_ * HKV_ * D_ * 4);
  float*  kp  = (float*)alloc((size_t)B_ * NB_ * HKV_ * D_ * 4);
  int*    msk = (int*)alloc((size_t)B_ * HKV_ * NB_ * NB_ * 4);
  __bf16* abf = (__bf16*)alloc(M * H_ * D_ * 2);
  if (off > ws_size) return;

  // 1) casts to bf16
  cast_bf16_kernel<<<4096, 256, 0, stream>>>(hidden, hbf, M * HID_);
  cast_bf16_kernel<<<4096, 256, 0, stream>>>(Wq, wqb, (size_t)HID_ * H_ * D_);
  cast_bf16_kernel<<<4096, 256, 0, stream>>>(Wk, wkb, (size_t)HID_ * HKV_ * D_);
  cast_bf16_kernel<<<4096, 256, 0, stream>>>(Wv, wvb, (size_t)HID_ * HKV_ * D_);
  cast_bf16_kernel<<<4096, 256, 0, stream>>>(Wo, wob, (size_t)H_ * D_ * HID_);

  // 2) projections
  {
    dim3 g((H_ * D_) / 64, (int)(M / 128));
    gemm_bf16_kernel<<<g, 256, 0, stream>>>(hbf, wqb, Qf, (int)M, H_ * D_, HID_);
  }
  {
    dim3 g((HKV_ * D_) / 64, (int)(M / 128));
    gemm_bf16_kernel<<<g, 256, 0, stream>>>(hbf, wkb, Kf, (int)M, HKV_ * D_, HID_);
    gemm_bf16_kernel<<<g, 256, 0, stream>>>(hbf, wvb, Vf, (int)M, HKV_ * D_, HID_);
  }

  // 3) RMSNorm + RoPE -> bf16 q/k ; cast v
  normrope_kernel<<<B_ * S_ * H_, 128, 0, stream>>>(Qf, qn, cosb, sinb, qbf, H_);
  normrope_kernel<<<B_ * S_ * HKV_, 128, 0, stream>>>(Kf, kn, cosb, sinb, kbf, HKV_);
  cast_bf16_kernel<<<4096, 256, 0, stream>>>(Vf, vbf, M * HKV_ * D_);

  // 4) block-sparse mask
  pool_kernel<<<B_ * NB_ * HKV_, 128, 0, stream>>>(qbf, kbf, qp, kp);
  blockmask_kernel<<<B_ * HKV_ * NB_, 32, 0, stream>>>(qp, kp, msk);

  // 5) attention
  attn_wmma_kernel<<<B_ * H_ * NB_, 128, 0, stream>>>(qbf, kbf, vbf, msk, abf);

  // 6) output projection -> d_out
  {
    dim3 g(HID_ / 64, (int)(M / 128));
    gemm_bf16_kernel<<<g, 256, 0, stream>>>(abf, wob, out, (int)M, HID_, H_ * D_);
  }
}
